// GAT_77627238908082
// MI455X (gfx1250) — compile-verified, matrix operand order checked by reading, not run on hardware
//
#include <hip/hip_runtime.h>
#include <stdint.h>
#include <math.h>

// GAT (3-layer, 4 heads, mean-aggregated) for MI455X / gfx1250.
// - Projections: v_wmma_f32_16x16x32_f16, fed from pre-converted f16 activations
//   and a per-layer pre-swizzled f16 W laid out in exact B-fragment order
//   (one contiguous 32B read per lane per K-step; zero in-loop cvt).
// - Edge softmax/aggregation: f32 atomics in L2 (h16/num ~ 38MB << 192MB L2);
//   feature gather reads f16 to halve the dominant L2 read traffic.

#define GAT_HEADS 4
#define GAT_C     32          // per-head channels (F_HID == F_OUT)
#define GAT_HC    128         // HEADS * GAT_C
#define GAT_FIN   128
#define NEG_SLOPE 0.2f

typedef __attribute__((ext_vector_type(16))) _Float16 v16h;
typedef __attribute__((ext_vector_type(8)))  _Float16 v8h;
typedef __attribute__((ext_vector_type(8)))  float    v8f;

// float atomic max via signed/unsigned integer atomics (valid for mixed signs,
// initialized to -inf). Lowers to GLOBAL_ATOMIC_MAX_I32 / MIN_U32 in L2.
__device__ __forceinline__ void atomic_max_f32(float* addr, float v) {
  if (v >= 0.0f) atomicMax((int*)addr, __float_as_int(v));
  else           atomicMin((unsigned int*)addr, __float_as_uint(v));
}

// f32 -> f16 bulk convert (used once for the input features)
__global__ void cvt_f32_to_f16(const float* __restrict__ src,
                               _Float16* __restrict__ dst, long long count) {
  long long i = (long long)blockIdx.x * blockDim.x + threadIdx.x;
  if (i < count) dst[i] = (_Float16)src[i];
}

// Pre-swizzle W[K,128] (f32, row-major) into WMMA B-fragment order, f16:
//   Wswz[((colTile*(K/32) + kblk)*32 + lane)*16 + e]
// where element e of lane L covers K = kblk*32 + (e<8 ? ko+e : 16+ko+e-8),
// ko = (L&16)>>1, column = colTile*16 + (L&15).
__global__ void gat_swizzle_W(const float* __restrict__ W,
                              _Float16* __restrict__ Wswz, int K) {
  int idx = blockIdx.x * blockDim.x + threadIdx.x;
  if (idx >= K * GAT_HC) return;
  int kb32 = K >> 5;
  int e    = idx & 15;
  int lane = (idx >> 4) & 31;
  int t    = idx >> 9;
  int kblk = t % kb32;
  int ct   = t / kb32;
  int ko   = (lane & 16) >> 1;
  int k    = kblk * 32 + ((e < 8) ? (ko + e) : (16 + ko + (e - 8)));
  int col  = ct * 16 + (lane & 15);
  Wswz[idx] = (_Float16)W[(size_t)k * GAT_HC + col];
}

// ---------------------------------------------------------------------------
// h16[N,128] = (f16) A16[N,K] @ W  — one wave per 16x16 output tile.
// A fragment: lane L holds row (L&15); two contiguous 8-half runs at K offsets
// ko and ko+16 per 32-wide K step (ko = (L&16)>>1).  B from Wswz: one
// contiguous v16h (32B) per lane per K step.  D: VGPR r -> M = r+ko, N = L&15.
// ---------------------------------------------------------------------------
template <int K>
__launch_bounds__(256)
__global__ void gat_wmma_gemm(const _Float16* __restrict__ A,
                              const _Float16* __restrict__ Wswz,
                              _Float16* __restrict__ H16, int n) {
  constexpr int KB32 = K >> 5;
  const int lane = threadIdx.x & 31;
  const int wave = threadIdx.x >> 5;           // 8 waves -> 8 column tiles
  const int rt   = blockIdx.x;                 // 16-row tile
  const int m    = lane & 15;
  const int ko   = (lane & 16) >> 1;           // 0 or 8
  const int ncol = wave * 16 + m;

  int row = rt * 16 + m;
  if (row >= n) row = n - 1;                   // N = 50000 is 16-aligned; safety clamp
  const _Float16* arow = A + (size_t)row * K;
  const _Float16* bptr = Wswz + (((size_t)wave * KB32) * 32 + lane) * 16;

  v8f acc = {};
#pragma unroll
  for (int kblk = 0; kblk < KB32; ++kblk) {
    v8h r0 = *(const v8h*)(arow + kblk * 32 + ko);
    v8h r1 = *(const v8h*)(arow + kblk * 32 + ko + 16);
    v16h a;
#pragma unroll
    for (int j = 0; j < 8; ++j) { a[j] = r0[j]; a[j + 8] = r1[j]; }
    v16h b = *(const v16h*)(bptr + (size_t)kblk * 32 * 16);
    acc = __builtin_amdgcn_wmma_f32_16x16x32_f16(false, a, false, b,
                                                 (short)0, acc, false, false);
  }

  // Epilogue: wave-uniform bounds test; full tiles (the common case — N is a
  // multiple of 16) take the unpredicated path so the 8 b16 stores clause up.
  _Float16* hout = H16 + ((size_t)(rt * 16 + ko)) * GAT_HC + ncol;
  if (rt * 16 + 16 <= n) {
#pragma unroll
    for (int r = 0; r < 8; ++r) hout[(size_t)r * GAT_HC] = (_Float16)acc[r];
  } else {
#pragma unroll
    for (int r = 0; r < 8; ++r)
      if (rt * 16 + ko + r < n) hout[(size_t)r * GAT_HC] = (_Float16)acc[r];
  }
}

// score_{s,d}[n,h] = sum_c h[n,h,c] * a_{src,dst}[h,c]
__global__ void gat_scores(const _Float16* __restrict__ H16,
                           const float* __restrict__ a_src,
                           const float* __restrict__ a_dst,
                           float* __restrict__ sS, float* __restrict__ sD, int n) {
  int i = blockIdx.x * blockDim.x + threadIdx.x;
  if (i >= n * GAT_HEADS) return;
  int node = i >> 2, hh = i & 3;
  const _Float16* hp = H16 + (size_t)node * GAT_HC + hh * GAT_C;
  const float* as = a_src + hh * GAT_C;
  const float* ad = a_dst + hh * GAT_C;
  float ss = 0.f, sd = 0.f;
#pragma unroll
  for (int c = 0; c < GAT_C; ++c) {
    float hv = (float)hp[c];
    ss += hv * as[c];
    sd += hv * ad[c];
  }
  sS[i] = ss; sD[i] = sd;
}

// zero numerator, -inf running max, zero denominator (re-run every call)
__global__ void gat_init(float* __restrict__ num, float* __restrict__ m,
                         float* __restrict__ denom, int n) {
  int i = blockIdx.x * blockDim.x + threadIdx.x;
  if (i < n * GAT_HC) num[i] = 0.0f;
  if (i < n * GAT_HEADS) { m[i] = -INFINITY; denom[i] = 0.0f; }
}

// pass 1: e = leaky_relu(sS[src]+sD[dst]); segment max into m[dst]
__global__ void gat_edge_max(const long long* __restrict__ ei, long long E, long long tot,
                             const float* __restrict__ sS, const float* __restrict__ sD,
                             float* __restrict__ ebuf, float* __restrict__ m) {
  long long idx = (long long)blockIdx.x * blockDim.x + threadIdx.x;
  if (idx >= tot) return;
  long long s, d;
  if (idx < E) { s = ei[idx]; d = ei[E + idx]; } else { s = d = idx - E; }  // self-loops
#pragma unroll
  for (int hh = 0; hh < GAT_HEADS; ++hh) {
    float ev = sS[s * GAT_HEADS + hh] + sD[d * GAT_HEADS + hh];
    ev = ev > 0.0f ? ev : NEG_SLOPE * ev;
    ebuf[idx * GAT_HEADS + hh] = ev;
    atomic_max_f32(&m[d * GAT_HEADS + hh], ev);
  }
}

// pass 2: ex = exp(e - m[dst]); num[dst,h,:] += h[src,h,:]*ex; denom[dst,h] += ex
// one wave per (edge, head); lane = channel.  Gather is f16 (halves L2 traffic),
// accumulation stays f32 (GLOBAL_ATOMIC_ADD_F32).
__launch_bounds__(256)
__global__ void gat_edge_acc(const long long* __restrict__ ei, long long E, long long tot,
                             const float* __restrict__ ebuf, const float* __restrict__ m,
                             const _Float16* __restrict__ H16,
                             float* __restrict__ num, float* __restrict__ denom) {
  long long wid  = (long long)blockIdx.x * 8 + (threadIdx.x >> 5);
  int       lane = threadIdx.x & 31;
  long long edge = wid >> 2;
  int       hh   = (int)(wid & 3);
  if (edge >= tot) return;
  long long s, d;
  if (edge < E) { s = ei[edge]; d = ei[E + edge]; } else { s = d = edge - E; }
  float mv = m[d * GAT_HEADS + hh];
  if (!__builtin_isfinite(mv)) mv = 0.0f;        // matches reference isfinite guard
  float ex = expf(ebuf[edge * GAT_HEADS + hh] - mv);
  float hv = (float)H16[s * GAT_HC + hh * GAT_C + lane];
  atomicAdd(&num[d * GAT_HC + hh * GAT_C + lane], hv * ex);
  if (lane == 0) atomicAdd(&denom[d * GAT_HEADS + hh], ex);
}

// out[n,c] = mean_h( num[n,h,c] / (denom[n,h]+1e-16) ) + b[c]
// Writes f32 (final layer) and/or f16 (feeds next layer's GEMM) per pointers.
__global__ void gat_finalize(const float* __restrict__ num, const float* __restrict__ denom,
                             const float* __restrict__ bias,
                             float* __restrict__ out, _Float16* __restrict__ out16, int n) {
  int i = blockIdx.x * blockDim.x + threadIdx.x;
  if (i >= n * GAT_C) return;
  int node = i >> 5, c = i & 31;
  float acc = 0.0f;
#pragma unroll
  for (int hh = 0; hh < GAT_HEADS; ++hh)
    acc += num[(size_t)node * GAT_HC + hh * GAT_C + c] /
           (denom[node * GAT_HEADS + hh] + 1e-16f);
  float r = acc * (1.0f / GAT_HEADS) + bias[c];
  if (out)   out[(size_t)node * GAT_C + c]   = r;
  if (out16) out16[(size_t)node * GAT_C + c] = (_Float16)r;
}

// ---------------------------------------------------------------------------
namespace {
struct Ws {
  _Float16 *xh, *h16, *hmid16, *wswz;
  float *num, *sS, *sD, *m, *denom, *ebuf;
};

template <int K>
void run_layer(const _Float16* xin16, const float* W, const float* as, const float* ad,
               const float* bias, const long long* ei, long long E, int n,
               const Ws& w, float* out, _Float16* out16, hipStream_t stream) {
  long long tot = E + n;
  gat_swizzle_W<<<(K * GAT_HC + 255) / 256, 256, 0, stream>>>(W, w.wswz, K);
  gat_wmma_gemm<K><<<dim3((n + 15) / 16), 256, 0, stream>>>(xin16, w.wswz, w.h16, n);
  gat_scores<<<(n * GAT_HEADS + 255) / 256, 256, 0, stream>>>(w.h16, as, ad, w.sS, w.sD, n);
  gat_init<<<(n * GAT_HC + 255) / 256, 256, 0, stream>>>(w.num, w.m, w.denom, n);
  gat_edge_max<<<(unsigned)((tot + 255) / 256), 256, 0, stream>>>(ei, E, tot, w.sS, w.sD,
                                                                  w.ebuf, w.m);
  long long waves = tot * GAT_HEADS;
  gat_edge_acc<<<(unsigned)((waves + 7) / 8), 256, 0, stream>>>(ei, E, tot, w.ebuf, w.m,
                                                                w.h16, w.num, w.denom);
  gat_finalize<<<(n * GAT_C + 255) / 256, 256, 0, stream>>>(w.num, w.denom, bias,
                                                            out, out16, n);
}
}  // namespace

extern "C" void kernel_launch(void* const* d_in, const int* in_sizes, int n_in,
                              void* d_out, int out_size, void* d_ws, size_t ws_size,
                              hipStream_t stream) {
  const float*     x  = (const float*)d_in[0];
  const long long* ei = (const long long*)d_in[1];   // edge_index is int64 in reference
  const float *W1 = (const float*)d_in[2],  *as1 = (const float*)d_in[3],
              *ad1 = (const float*)d_in[4], *b1  = (const float*)d_in[5];
  const float *W2 = (const float*)d_in[6],  *as2 = (const float*)d_in[7],
              *ad2 = (const float*)d_in[8], *b2  = (const float*)d_in[9];
  const float *W3 = (const float*)d_in[10], *as3 = (const float*)d_in[11],
              *ad3 = (const float*)d_in[12], *b3 = (const float*)d_in[13];

  const int n = in_sizes[0] / GAT_FIN;               // 50000
  const long long E = (long long)in_sizes[1] / 2;    // 800000
  const long long tot = E + n;

  char* base = (char*)d_ws;
  size_t off = 0;
  auto take = [&](size_t bytes) -> void* {
    void* p = (void*)(base + off);
    off = (off + bytes + 255) & ~(size_t)255;
    return p;
  };
  Ws w;
  w.xh     = (_Float16*)take((size_t)n * GAT_FIN * sizeof(_Float16));   // 12.8 MB
  w.h16    = (_Float16*)take((size_t)n * GAT_HC * sizeof(_Float16));    // 12.8 MB
  w.hmid16 = (_Float16*)take((size_t)n * GAT_C * sizeof(_Float16));     //  3.2 MB
  w.wswz   = (_Float16*)take((size_t)GAT_FIN * GAT_HC * sizeof(_Float16)); // 32 KB
  w.num    = (float*)take((size_t)n * GAT_HC * sizeof(float));          // 25.6 MB
  w.sS     = (float*)take((size_t)n * GAT_HEADS * sizeof(float));
  w.sD     = (float*)take((size_t)n * GAT_HEADS * sizeof(float));
  w.m      = (float*)take((size_t)n * GAT_HEADS * sizeof(float));
  w.denom  = (float*)take((size_t)n * GAT_HEADS * sizeof(float));
  w.ebuf   = (float*)take((size_t)tot * GAT_HEADS * sizeof(float));     // 13.6 MB
  if (off > ws_size) return;  // workspace too small — nothing safe to do

  // One-time f32->f16 conversion of the input features.
  long long xcnt = (long long)n * GAT_FIN;
  cvt_f32_to_f16<<<(unsigned)((xcnt + 255) / 256), 256, 0, stream>>>(x, w.xh, xcnt);

  // Layer 1: xh[N,128] -> hmid16[N,32] (f16 feeds layer 2)
  run_layer<GAT_FIN>(w.xh, W1, as1, ad1, b1, ei, E, n, w, nullptr, w.hmid16, stream);
  // Layer 2: hmid16 -> hmid16 (xin only read by the GEMM, which precedes finalize)
  run_layer<GAT_C>(w.hmid16, W2, as2, ad2, b2, ei, E, n, w, nullptr, w.hmid16, stream);
  // Layer 3: hmid16 -> d_out (f32)
  run_layer<GAT_C>(w.hmid16, W3, as3, ad3, b3, ei, E, n, w, (float*)d_out, nullptr, stream);
}
